// Encoder_Spatio_10050223473239
// MI455X (gfx1250) — matrix operand order repaired and checked by reading, not compiled, not run on hardware
//
#include <hip/hip_runtime.h>
#include <hip/hip_bf16.h>
#include <math.h>

// ---------------------------------------------------------------------------
// Problem constants: B=8, N=2048, NIN=NHID=256, D=64.  M_total = B*N = 16384.
// ---------------------------------------------------------------------------
#define MTOT   16384
#define NSEQ   2048
#define KDIM   256
#define DDIM   64

typedef __bf16 bf16_t;
typedef __attribute__((ext_vector_type(16))) __bf16 v16bf;
typedef __attribute__((ext_vector_type(8)))  __bf16 v8bf;
typedef __attribute__((ext_vector_type(4)))  __bf16 v4bf;
typedef __attribute__((ext_vector_type(8)))  float  v8f;
typedef __attribute__((ext_vector_type(4)))  float  v4f;

// Branchless ELU(alpha=1): x>0 ? x : exp(x)-1, using the hardware v_exp_f32.
__device__ __forceinline__ float elu1f(float x) {
  float e = __expf(fminf(x, 0.0f)) - 1.0f;
  return x > 0.0f ? x : e;
}

// ---------------------------------------------------------------------------
// Weight f32 -> bf16 conversion (tiny, once per call into workspace)
// ---------------------------------------------------------------------------
__global__ void __launch_bounds__(256)
cvt_f32_bf16(const float* __restrict__ s, bf16_t* __restrict__ d, int n) {
  int i = blockIdx.x * 256 + threadIdx.x;
  if (i < n) d[i] = (bf16_t)s[i];
}

// ---------------------------------------------------------------------------
// Fused per-row pipeline: inputs -> fc1+elu -> fc2+elu -> BN -> q/k proj ->
// sq/sk scalars.  One wave32 per 16-row tile.  All GEMMs via bf16 WMMA.
// ---------------------------------------------------------------------------
__global__ void __launch_bounds__(32)
fused_rows(const float*  __restrict__ in,      // (16384, 256) f32
           const bf16_t* __restrict__ w1,      // (256, 256)  bf16, [h][i]
           const float*  __restrict__ b1,      // (256)
           const bf16_t* __restrict__ w2,      // (256, 256)  bf16
           const float*  __restrict__ b2,      // (256)
           const float*  __restrict__ bn_g,
           const float*  __restrict__ bn_bb,
           const float*  __restrict__ bn_mean,
           const float*  __restrict__ bn_var,
           const bf16_t* __restrict__ wq,      // (64, 256) bf16
           const float*  __restrict__ wq_b,    // (64)
           const bf16_t* __restrict__ wk,      // (64, 256) bf16
           const float*  __restrict__ wk_b,    // (64)
           const float*  __restrict__ mlp2_w,  // (128): [0:64]=wk_half, [64:128]=wq_half
           float*        __restrict__ sq,      // (16384)
           float*        __restrict__ sk)      // (16384)
{
  __shared__ bf16_t bufX[16 * KDIM];   // activation staging (bf16, row-major)
  __shared__ bf16_t bufY[16 * KDIM];
  __shared__ float  redQ[16];
  __shared__ float  redK[16];

  const int lane = threadIdx.x;        // 0..31
  const int m0   = blockIdx.x * 16;    // base row of this tile

  // ---- stage 0: cooperative load of 16x256 input tile, f32 -> bf16 in LDS
  for (int it = 0; it < 32; ++it) {
    int e4  = it * 32 + lane;          // 0..1023
    int idx = e4 * 4;
    int row = idx >> 8;
    int col = idx & 255;
    float4 v = *(const float4*)(in + (size_t)(m0 + row) * KDIM + col);
    v4bf p;
    p[0] = (bf16_t)v.x; p[1] = (bf16_t)v.y; p[2] = (bf16_t)v.z; p[3] = (bf16_t)v.w;
    *(v4bf*)(&bufX[idx]) = p;
  }
  if (lane < 16) { redQ[lane] = 0.0f; redK[lane] = 0.0f; }

  // Per-lane WMMA fragment geometry (wave32, 16x16x32 bf16).
  const int rowA  = lane & 15;               // A: lane -> M row
  const int kselA = (lane < 16) ? 0 : 8;     // A: halves 0..7 = K ksel..ksel+7, 8..15 = +16
  const int colB  = lane & 15;               // B: lane -> N col
  const int kselB = (lane < 16) ? 0 : 8;     // B: halves 0..7 = K ksel..ksel+7, 8..15 = +16
  const int cM    = (lane < 16) ? 0 : 8;     // C: VGPR r -> M = r + cM, N = lane&15

  // A fragment from a bf16 LDS activation buffer at K-tile kt
  auto loadA = [&](const bf16_t* buf, int kt) -> v16bf {
    const bf16_t* p = buf + rowA * KDIM + kt * 32 + kselA;
    v8bf lo = *(const v8bf*)(p);
    v8bf hi = *(const v8bf*)(p + 16);
    v16bf a;
#pragma unroll
    for (int i = 0; i < 8; ++i) { a[i] = lo[i]; a[8 + i] = hi[i]; }
    return a;
  };

  // B fragment from bf16 weights W (layout W[n][k], ld = 256): B(k,n) = W[n*256+k]
  auto loadB = [&](const bf16_t* W, int nbase, int kt) -> v16bf {
    const bf16_t* p = W + (size_t)(nbase + colB) * KDIM + kt * 32 + kselB;
    v8bf lo = *(const v8bf*)(p);
    v8bf hi = *(const v8bf*)(p + 16);
    v16bf b;
#pragma unroll
    for (int i = 0; i < 8; ++i) { b[i] = lo[i]; b[8 + i] = hi[i]; }
    return b;
  };

  v16bf afr[8];                        // full K-slice of A: 8 fragments = 64 VGPRs

  // ---- stage 1: x1 = elu(in @ W1^T + b1)  -> bufY (bf16)
#pragma unroll
  for (int kt = 0; kt < 8; ++kt) afr[kt] = loadA(bufX, kt);
  for (int nt = 0; nt < 16; ++nt) {
    v8f c;
#pragma unroll
    for (int i = 0; i < 8; ++i) c[i] = 0.0f;
#pragma unroll
    for (int kt = 0; kt < 8; ++kt) {
      v16bf b = loadB(w1, nt * 16, kt);
      c = __builtin_amdgcn_wmma_f32_16x16x32_bf16(false, afr[kt], false, b,
                                                  (short)0, c, false, false);
    }
    int   n    = nt * 16 + (lane & 15);
    float bias = b1[n];
#pragma unroll
    for (int r = 0; r < 8; ++r) {
      float v = elu1f(c[r] + bias);
      bufY[(r + cM) * KDIM + n] = (bf16_t)v;
    }
  }

  // ---- stage 2: X = BN(elu(x1 @ W2^T + b2)) -> bufX (bf16)
#pragma unroll
  for (int kt = 0; kt < 8; ++kt) afr[kt] = loadA(bufY, kt);
  for (int nt = 0; nt < 16; ++nt) {
    v8f c;
#pragma unroll
    for (int i = 0; i < 8; ++i) c[i] = 0.0f;
#pragma unroll
    for (int kt = 0; kt < 8; ++kt) {
      v16bf b = loadB(w2, nt * 16, kt);
      c = __builtin_amdgcn_wmma_f32_16x16x32_bf16(false, afr[kt], false, b,
                                                  (short)0, c, false, false);
    }
    int   n     = nt * 16 + (lane & 15);
    float bias  = b2[n];
    float scale = bn_g[n] * rsqrtf(bn_var[n] + 1e-5f);
    float mu    = bn_mean[n];
    float beta  = bn_bb[n];
#pragma unroll
    for (int r = 0; r < 8; ++r) {
      float v = elu1f(c[r] + bias);
      v = (v - mu) * scale + beta;
      bufX[(r + cM) * KDIM + n] = (bf16_t)v;
    }
  }

  // ---- stage 3: q = X @ Wq^T + bq ; k = X @ Wk^T + bk ; fold with mlp2_w
  // sq[m] = sum_d q[m,d] * mlp2_w[64+d] ; sk[m] = sum_d k[m,d] * mlp2_w[d]
#pragma unroll
  for (int kt = 0; kt < 8; ++kt) afr[kt] = loadA(bufX, kt);

  float pq[8], pk[8];
#pragma unroll
  for (int r = 0; r < 8; ++r) { pq[r] = 0.0f; pk[r] = 0.0f; }

  for (int nt = 0; nt < 4; ++nt) {            // D = 64 = 4 n-tiles
    v8f cq, ck;
#pragma unroll
    for (int i = 0; i < 8; ++i) { cq[i] = 0.0f; ck[i] = 0.0f; }
#pragma unroll
    for (int kt = 0; kt < 8; ++kt) {
      v16bf bq = loadB(wq, nt * 16, kt);
      v16bf bk = loadB(wk, nt * 16, kt);
      cq = __builtin_amdgcn_wmma_f32_16x16x32_bf16(false, afr[kt], false, bq,
                                                   (short)0, cq, false, false);
      ck = __builtin_amdgcn_wmma_f32_16x16x32_bf16(false, afr[kt], false, bk,
                                                   (short)0, ck, false, false);
    }
    int   n   = nt * 16 + (lane & 15);
    float qb  = wq_b[n];
    float kb  = wk_b[n];
    float whq = mlp2_w[DDIM + n];             // wq_half
    float whk = mlp2_w[n];                    // wk_half
#pragma unroll
    for (int r = 0; r < 8; ++r) {
      pq[r] += (cq[r] + qb) * whq;
      pk[r] += (ck[r] + kb) * whk;
    }
  }

  // cross-lane reduction over N via LDS float atomics (ds_add_f32)
#pragma unroll
  for (int r = 0; r < 8; ++r) {
    atomicAdd(&redQ[r + cM], pq[r]);
    atomicAdd(&redK[r + cM], pk[r]);
  }

  if (lane < 16) {
    sq[m0 + lane] = redQ[lane];
    sk[m0 + lane] = redK[lane];
  }
}

// ---------------------------------------------------------------------------
// probs[b,i,j] = sq[b,i] + sk[b,j] + mlp2_b.  Pure streaming: 128 MiB writes.
// One block per output row (16384 rows of 2048 f32), non-temporal b128 stores
// so the streamed output doesn't evict the L2 working set.
// ---------------------------------------------------------------------------
__global__ void __launch_bounds__(256)
probs_kernel(const float* __restrict__ sq, const float* __restrict__ sk,
             const float* __restrict__ mlp2_b, float* __restrict__ out)
{
  int r  = blockIdx.x;                 // global row in [0, 16384)
  int bN = (r >> 11) << 11;            // b * 2048
  float sv = sq[r] + mlp2_b[0];
  const v4f* skv = (const v4f*)(sk + bN);
  v4f*       ov  = (v4f*)(out + (size_t)r * NSEQ);
#pragma unroll 2
  for (int t = threadIdx.x; t < NSEQ / 4; t += 256) {
    v4f k = skv[t];
    v4f o;
    o.x = sv + k.x; o.y = sv + k.y; o.z = sv + k.z; o.w = sv + k.w;
    __builtin_nontemporal_store(o, ov + t);
  }
}

// ---------------------------------------------------------------------------
// Launch
// ---------------------------------------------------------------------------
extern "C" void kernel_launch(void* const* d_in, const int* in_sizes, int n_in,
                              void* d_out, int out_size, void* d_ws, size_t ws_size,
                              hipStream_t stream) {
  const float* in      = (const float*)d_in[0];
  const float* fc1_w   = (const float*)d_in[1];
  const float* fc1_b   = (const float*)d_in[2];
  const float* fc2_w   = (const float*)d_in[3];
  const float* fc2_b   = (const float*)d_in[4];
  const float* bn_g    = (const float*)d_in[5];
  const float* bn_b    = (const float*)d_in[6];
  const float* bn_mean = (const float*)d_in[7];
  const float* bn_var  = (const float*)d_in[8];
  const float* wq_w    = (const float*)d_in[9];
  const float* wq_b    = (const float*)d_in[10];
  const float* wk_w    = (const float*)d_in[11];
  const float* wk_b    = (const float*)d_in[12];
  const float* mlp2_w  = (const float*)d_in[13];
  const float* mlp2_b  = (const float*)d_in[14];

  // Workspace layout (bytes):
  //  [0,       131072) fc1_w bf16 (65536)
  //  [131072,  262144) fc2_w bf16 (65536)
  //  [262144,  294912) wq_w  bf16 (16384)
  //  [294912,  327680) wk_w  bf16 (16384)
  //  [327680,  393216) sq f32 (16384)
  //  [393216,  458752) sk f32 (16384)
  char* ws = (char*)d_ws;
  bf16_t* w1b = (bf16_t*)(ws);
  bf16_t* w2b = (bf16_t*)(ws + 131072);
  bf16_t* wqb = (bf16_t*)(ws + 262144);
  bf16_t* wkb = (bf16_t*)(ws + 294912);
  float*  sq  = (float*)(ws + 327680);
  float*  sk  = (float*)(ws + 393216);

  cvt_f32_bf16<<<(65536 + 255) / 256, 256, 0, stream>>>(fc1_w, w1b, 65536);
  cvt_f32_bf16<<<(65536 + 255) / 256, 256, 0, stream>>>(fc2_w, w2b, 65536);
  cvt_f32_bf16<<<(16384 + 255) / 256, 256, 0, stream>>>(wq_w, wqb, 16384);
  cvt_f32_bf16<<<(16384 + 255) / 256, 256, 0, stream>>>(wk_w, wkb, 16384);

  fused_rows<<<MTOT / 16, 32, 0, stream>>>(in, w1b, fc1_b, w2b, fc2_b,
                                           bn_g, bn_b, bn_mean, bn_var,
                                           wqb, wq_b, wkb, wk_b,
                                           mlp2_w, sq, sk);

  probs_kernel<<<MTOT, 256, 0, stream>>>(sq, sk, mlp2_b, (float*)d_out);
}
